// BatchedImplicitCore_76794015252984
// MI455X (gfx1250) — matrix operand order repaired and checked by reading, not compiled, not run on hardware
//
#include <hip/hip_runtime.h>
#include <hip/hip_bf16.h>
#include <math.h>

// ---------------------------------------------------------------------------
// Implicit GNN fixed-point solver for MI455X (gfx1250), wave32 + WMMA bf16.
//   B=64, N=512, H=256, 30 damped iterations.  ~390 GFLOP, working set
//   (~134MB) fits in the 192MB L2 -> compute bound -> push all GEMMs through
//   v_wmma_f32_16x16x32_bf16 (f32 accumulate), feed operands with
//   ds_load_b128 fragments and stage panels with async global->LDS DMA.
// ---------------------------------------------------------------------------

typedef __bf16 bf16_t;
typedef __attribute__((ext_vector_type(16))) __bf16 v16bf;
typedef __attribute__((ext_vector_type(8)))  __bf16 v8bf;
typedef __attribute__((ext_vector_type(8)))  float  v8f;

#define BB    64
#define NN    512
#define HH    256
#define MTI   64          // iteration kernel: rows per workgroup tile
#define MTP   32          // pre kernel: rows per workgroup tile
#define NITER 30
#define PSTR  40          // padded k-stride of a transposed panel (elements)
#define PANEL_ELEMS (HH * PSTR)            // 10240 bf16 = 20480 B
#define PANEL_U4    (PANEL_ELEMS * 2 / 16) // 1280 uint4 chunks (5 per thread)

#if __has_builtin(__builtin_amdgcn_global_load_async_to_lds_b128)
#define ASYNC_COPY 1
typedef __attribute__((ext_vector_type(4))) int v4i;
typedef __attribute__((address_space(1))) v4i* gv4i_p;   // global v4i*
typedef __attribute__((address_space(3))) v4i* lv4i_p;   // LDS v4i*
#else
#define ASYNC_COPY 0
#endif

// Cooperative global -> LDS copy of NCHUNK uint4 per thread (256 threads).
// Async path: GLOBAL_LOAD_ASYNC_TO_LDS_B128 (ASYNCcnt DMA, no VGPR staging).
template <int NCHUNK>
__device__ __forceinline__ void copy_g2s(bf16_t* s_dst, const bf16_t* g_src,
                                         int tid) {
#if ASYNC_COPY
  char* g = (char*)const_cast<bf16_t*>(g_src) + tid * 16;
  char* s = (char*)s_dst + tid * 16;
#pragma unroll
  for (int j = 0; j < NCHUNK; ++j)
    __builtin_amdgcn_global_load_async_to_lds_b128(
        (gv4i_p)(g + j * 4096), (lv4i_p)(s + j * 4096), 0, 0);
#else
  const uint4* g = (const uint4*)g_src;
  uint4* s = (uint4*)s_dst;
  uint4 tmp[NCHUNK];
#pragma unroll
  for (int j = 0; j < NCHUNK; ++j) tmp[j] = g[tid + 256 * j];
#pragma unroll
  for (int j = 0; j < NCHUNK; ++j) s[tid + 256 * j] = tmp[j];
#endif
}

__device__ __forceinline__ void wait_async() {
#if ASYNC_COPY
#if __has_builtin(__builtin_amdgcn_s_wait_asynccnt)
  __builtin_amdgcn_s_wait_asynccnt(0);
#else
  asm volatile("s_wait_asynccnt 0x0" ::: "memory");
#endif
#endif
}

// ---------------------------------------------------------------------------
// Fragment loaders (LDS -> VGPR), ISA 7.12.2 layouts, wave32.
// ---------------------------------------------------------------------------
// A (16x32 bf16) from row-major [rows][ldk] LDS: lane L = row L%16;
// lanes<16 hold K {0..7, 16..23}, lanes>=16 hold K {8..15, 24..31}.
__device__ __forceinline__ v16bf frag_a(const bf16_t* s, int ldk,
                                        int rowbase, int kbase, int lane) {
  const int l16 = lane & 15, lh = lane >> 4;
  const bf16_t* p = s + (size_t)(rowbase + l16) * ldk + kbase + lh * 8;
  v8bf lo = *(const v8bf*)(p);        // ds_load_b128
  v8bf hi = *(const v8bf*)(p + 16);   // ds_load_b128
  return __builtin_shufflevector(lo, hi, 0, 1, 2, 3, 4, 5, 6, 7,
                                 8, 9, 10, 11, 12, 13, 14, 15);
}

// B (32x16 bf16) from a transposed panel [256 cols][PSTR] in LDS:
// lane L = column L%16; lanes<16 hold K {0..7, 16..23}, lanes>=16 hold K {8..15, 24..31}.
__device__ __forceinline__ v16bf frag_b(const bf16_t* s, int colbase, int lane) {
  const int l16 = lane & 15, lh = lane >> 4;
  const bf16_t* p = s + (size_t)(colbase + l16) * PSTR + lh * 8;
  v8bf lo = *(const v8bf*)(p);        // ds_load_b128 (80B*col + 16B*lh aligned)
  v8bf hi = *(const v8bf*)(p + 16);
  return __builtin_shufflevector(lo, hi, 0, 1, 2, 3, 4, 5, 6, 7,
                                 8, 9, 10, 11, 12, 13, 14, 15);
}

#define WMMA_BF16(a, b, c) \
  __builtin_amdgcn_wmma_f32_16x16x32_bf16(false, (a), false, (b), (short)0, (c), false, false)

// ---------------------------------------------------------------------------
// K1: spectral-norm power iteration (5 iters) on W (256x256); emit panel-
// transposed bf16 W^T_proj and Omega^T: element (k=h, n=d) at
//   [ (h/32)*HH + d ]*PSTR + h%32
// ---------------------------------------------------------------------------
__global__ __launch_bounds__(256)
void spec_kernel(const float* __restrict__ W, const float* __restrict__ Om,
                 bf16_t* __restrict__ wtT, bf16_t* __restrict__ omtT) {
  __shared__ float su[HH], sv[HH], red[HH];
  const int t = threadIdx.x;
  su[t] = 0.0625f;               // 1/sqrt(256)
  __syncthreads();
  for (int it = 0; it < 5; ++it) {
    float v = 0.f;
    for (int i = 0; i < HH; ++i) v += W[i * HH + t] * su[i];     // v = W^T u
    red[t] = v * v; __syncthreads();
    for (int s = 128; s > 0; s >>= 1) { if (t < s) red[t] += red[t + s]; __syncthreads(); }
    float nv = sqrtf(red[0]); __syncthreads();
    sv[t] = v / (nv + 1e-12f);
    __syncthreads();
    float u = 0.f;
    for (int j = 0; j < HH; ++j) u += W[t * HH + j] * sv[j];     // u = W v
    red[t] = u * u; __syncthreads();
    for (int s = 128; s > 0; s >>= 1) { if (t < s) red[t] += red[t + s]; __syncthreads(); }
    float nu = sqrtf(red[0]); __syncthreads();
    su[t] = u / (nu + 1e-12f);
    __syncthreads();
  }
  float wv = 0.f;
  for (int j = 0; j < HH; ++j) wv += W[t * HH + j] * sv[j];
  red[t] = su[t] * wv; __syncthreads();
  for (int s = 128; s > 0; s >>= 1) { if (t < s) red[t] += red[t + s]; __syncthreads(); }
  const float sigma = red[0];
  const float scale = fminf(1.0f, 0.999f / (sigma + 1e-12f));
  for (int i = t; i < HH * HH; i += 256) {
    const int h = i >> 8, d = i & 255;
    const size_t o = (size_t)((h >> 5) * HH + d) * PSTR + (h & 31);
    wtT[o]  = (bf16_t)(W[d * HH + h] * scale);
    omtT[o] = (bf16_t)(Om[d * HH + h]);
  }
}

// ---------------------------------------------------------------------------
// K2: bf16 conversion of adj & Hfeat; zero-init Z0 (f32 in d_out) and the
//     panel-transposed bf16 Z mirror.
// ---------------------------------------------------------------------------
__global__ __launch_bounds__(256)
void conv_kernel(const float* __restrict__ adj, const float* __restrict__ Hf,
                 bf16_t* __restrict__ adjb, bf16_t* __restrict__ hfb,
                 float* __restrict__ zf0, bf16_t* __restrict__ zbT0) {
  const size_t stride = (size_t)gridDim.x * blockDim.x;
  const size_t gid = (size_t)blockIdx.x * blockDim.x + threadIdx.x;
  const size_t NADJ = (size_t)BB * NN * NN;
  const size_t NZ   = (size_t)BB * NN * HH;
  const size_t NZB  = (size_t)BB * (NN / 32) * PANEL_ELEMS;
  for (size_t i = gid; i < NADJ; i += stride) adjb[i] = (bf16_t)adj[i];
  for (size_t i = gid; i < NZ; i += stride) {
    hfb[i] = (bf16_t)Hf[i];
    zf0[i] = 0.0f;
  }
  for (size_t i = gid; i < NZB; i += stride) zbT0[i] = (bf16_t)0.0f;
}

// ---------------------------------------------------------------------------
// K3: pre = Hfeat * Omega^T + Q + bias   ([32768 x 256] x [256 x 256])
// One WG = 32 rows, 256 threads = 8 waves (2 row-groups x 4 col-groups).
// ---------------------------------------------------------------------------
__global__ __launch_bounds__(256)
void pre_kernel(const bf16_t* __restrict__ hfb, const bf16_t* __restrict__ omtT,
                const float* __restrict__ Q, const float* __restrict__ bias,
                float* __restrict__ pre) {
  extern __shared__ char smem[];
  bf16_t* s_a   = (bf16_t*)smem;                   // 32 x 256 A tile (16KB)
  bf16_t* s_pan = (bf16_t*)(smem + MTP * HH * 2);  // transposed panel (20KB)
  const int tid = threadIdx.x;
  const int lane = tid & 31, wave = tid >> 5;
  const int waveM = wave >> 2, waveN = wave & 3;
  const int m0 = blockIdx.x * MTP;

  copy_g2s<4>(s_a, hfb + (size_t)m0 * HH, tid);    // contiguous 32x256 A tile
  v8f c[4] = {};

  for (int kk = 0; kk < 8; ++kk) {
    copy_g2s<5>(s_pan, omtT + (size_t)kk * PANEL_ELEMS, tid);
    wait_async();
    __syncthreads();
    v16bf a = frag_a(s_a, HH, waveM * 16, kk * 32, lane);
#pragma unroll
    for (int t = 0; t < 4; ++t) {
      v16bf b = frag_b(s_pan, waveN * 64 + t * 16, lane);
      c[t] = WMMA_BF16(a, b, c[t]);
    }
    __syncthreads();
  }

  const int l16 = lane & 15, lh = lane >> 4;
#pragma unroll
  for (int t = 0; t < 4; ++t)
#pragma unroll
    for (int i = 0; i < 8; ++i) {
      const int r   = waveM * 16 + lh * 8 + i;
      const int col = waveN * 64 + t * 16 + l16;
      const size_t idx = (size_t)(m0 + r) * HH + col;
      pre[idx] = c[t][i] + Q[idx] + bias[col];
    }
}

// ---------------------------------------------------------------------------
// K4 (x30): one damped fixed-point step, fused per 64-row tile:
//   AZ = adj[tile,:] x Z   (K=512) ;  P = AZ x Wproj^T  (K=256)
//   Z' = 0.5 Z + 0.5 tanh(P + pre) * mask
// 8 waves, each owning 2 m-subtiles x 4 n-subtiles (8 accumulators); B
// fragments reused across m-subtiles; panels double-buffered with async DMA.
// ---------------------------------------------------------------------------
__global__ __launch_bounds__(256)
void iter_kernel(const bf16_t* __restrict__ adjb, const bf16_t* __restrict__ zbT_cur,
                 const float* __restrict__ zf_cur, const bf16_t* __restrict__ wtT,
                 const float* __restrict__ pre, const float* __restrict__ mask,
                 float* __restrict__ zf_next, bf16_t* __restrict__ zbT_next) {
  extern __shared__ char smem[];
  bf16_t* s_adj = (bf16_t*)smem;                                   // 64x512 = 64KB
  bf16_t* s_pbuf[2];
  s_pbuf[0] = (bf16_t*)(smem + MTI * NN * 2);                      // panel 20KB
  s_pbuf[1] = (bf16_t*)(smem + MTI * NN * 2 + PANEL_ELEMS * 2);    // panel 20KB
  bf16_t* s_az = (bf16_t*)(smem + MTI * NN * 2 + 2 * PANEL_ELEMS * 2); // 64x256 32KB

  const int b  = blockIdx.y;
  const int m0 = blockIdx.x * MTI;
  const int tid = threadIdx.x;
  const int lane = tid & 31, wave = tid >> 5;
  const int waveM = wave >> 2, waveN = wave & 3;
  const int l16 = lane & 15, lh = lane >> 4;

  const bf16_t* zbB = zbT_cur + (size_t)b * (NN / 32) * PANEL_ELEMS;

  // adjacency tile (contiguous 64x512 bf16) + first Z panel, async
  copy_g2s<16>(s_adj, adjb + ((size_t)b * NN + m0) * NN, tid);
  copy_g2s<5>(s_pbuf[0], zbB, tid);
  wait_async();
  __syncthreads();

  v8f c[2][4] = {};
  int cur = 0;

  // ---- GEMM1: AZ = adj_tile x Z, K = 512 in 16 transposed panels ----
  for (int kk = 0; kk < 16; ++kk) {
    const int nxt = cur ^ 1;
    if (kk + 1 < 16)
      copy_g2s<5>(s_pbuf[nxt], zbB + (size_t)(kk + 1) * PANEL_ELEMS, tid);
    else
      copy_g2s<5>(s_pbuf[nxt], wtT, tid);   // prefetch first W^T panel
    v16bf a0 = frag_a(s_adj, NN, waveM * 32,      kk * 32, lane);
    v16bf a1 = frag_a(s_adj, NN, waveM * 32 + 16, kk * 32, lane);
#pragma unroll
    for (int t = 0; t < 4; ++t) {
      v16bf bf = frag_b(s_pbuf[cur], waveN * 64 + t * 16, lane);
      c[0][t] = WMMA_BF16(a0, bf, c[0][t]);
      c[1][t] = WMMA_BF16(a1, bf, c[1][t]);
    }
    wait_async();
    __syncthreads();
    cur = nxt;
  }

  // stage AZ in LDS as bf16, row-major (A-operand for GEMM2)
#pragma unroll
  for (int ms = 0; ms < 2; ++ms)
#pragma unroll
    for (int t = 0; t < 4; ++t)
#pragma unroll
      for (int i = 0; i < 8; ++i)
        s_az[(size_t)(waveM * 32 + ms * 16 + lh * 8 + i) * HH +
             waveN * 64 + t * 16 + l16] = (bf16_t)c[ms][t][i];
  __syncthreads();

  // ---- GEMM2: P = AZ x Wproj^T, K = 256 in 8 transposed panels ----
  v8f zero = {};
#pragma unroll
  for (int ms = 0; ms < 2; ++ms)
#pragma unroll
    for (int t = 0; t < 4; ++t) c[ms][t] = zero;
  for (int kk = 0; kk < 8; ++kk) {
    const int nxt = cur ^ 1;
    if (kk + 1 < 8)
      copy_g2s<5>(s_pbuf[nxt], wtT + (size_t)(kk + 1) * PANEL_ELEMS, tid);
    v16bf a0 = frag_a(s_az, HH, waveM * 32,      kk * 32, lane);
    v16bf a1 = frag_a(s_az, HH, waveM * 32 + 16, kk * 32, lane);
#pragma unroll
    for (int t = 0; t < 4; ++t) {
      v16bf bf = frag_b(s_pbuf[cur], waveN * 64 + t * 16, lane);
      c[0][t] = WMMA_BF16(a0, bf, c[0][t]);
      c[1][t] = WMMA_BF16(a1, bf, c[1][t]);
    }
    wait_async();
    __syncthreads();
    cur = nxt;
  }

  // ---- epilogue: tanh + mask + damping; scatter Z' into both layouts ----
  const size_t rowBase = (size_t)b * NN + m0;
  bf16_t* zbT_tile = zbT_next +
      ((size_t)b * (NN / 32) + (size_t)blockIdx.x * 2) * PANEL_ELEMS;
#pragma unroll
  for (int ms = 0; ms < 2; ++ms)
#pragma unroll
    for (int t = 0; t < 4; ++t)
#pragma unroll
      for (int i = 0; i < 8; ++i) {
        const int r   = waveM * 32 + ms * 16 + lh * 8 + i;   // 0..63
        const int col = waveN * 64 + t * 16 + l16;
        const size_t idx = (rowBase + r) * HH + col;
        const float zn = tanhf(c[ms][t][i] + pre[idx]) * mask[b * NN + m0 + r];
        const float z  = 0.5f * zf_cur[idx] + 0.5f * zn;
        zf_next[idx] = z;
        zbT_tile[(size_t)(r >> 5) * PANEL_ELEMS + (size_t)col * PSTR + (r & 31)] =
            (bf16_t)z;
      }
}

// ---------------------------------------------------------------------------
// Host side. ws layout (256B aligned):
//   adjb 32MB | hfb 16MB | wtT 160KB | omtT 160KB | pre 32MB | zf1 32MB
//   | zbT0 20MB | zbT1 20MB   (~152MB total)
// f32 Z ping-pongs between ws.zf1 and d_out so iter 29 writes d_out.
// ---------------------------------------------------------------------------
extern "C" void kernel_launch(void* const* d_in, const int* in_sizes, int n_in,
                              void* d_out, int out_size, void* d_ws, size_t ws_size,
                              hipStream_t stream) {
  const float* Hf   = (const float*)d_in[0];
  const float* Q    = (const float*)d_in[1];
  const float* adj  = (const float*)d_in[2];
  const float* mask = (const float*)d_in[3];
  const float* W    = (const float*)d_in[4];
  const float* Om   = (const float*)d_in[5];
  const float* bias = (const float*)d_in[6];
  float* out = (float*)d_out;

  char* ws = (char*)d_ws;
  size_t off = 0;
  auto carve = [&](size_t bytes) -> char* {
    char* p = ws + off;
    off = (off + bytes + 255) & ~(size_t)255;
    return p;
  };
  const size_t NADJ = (size_t)BB * NN * NN;                  // 16.8M
  const size_t NZ   = (size_t)BB * NN * HH;                  // 8.4M
  const size_t NZB  = (size_t)BB * (NN / 32) * PANEL_ELEMS;  // 10.5M
  bf16_t* adjb = (bf16_t*)carve(NADJ * 2);
  bf16_t* hfb  = (bf16_t*)carve(NZ * 2);
  bf16_t* wtT  = (bf16_t*)carve((size_t)(HH / 32) * PANEL_ELEMS * 2);
  bf16_t* omtT = (bf16_t*)carve((size_t)(HH / 32) * PANEL_ELEMS * 2);
  float*  pre  = (float*)carve(NZ * 4);
  float*  zf1  = (float*)carve(NZ * 4);
  bf16_t* zbT0 = (bf16_t*)carve(NZB * 2);
  bf16_t* zbT1 = (bf16_t*)carve(NZB * 2);

  // K1: spectral projection + panel-transposed bf16 weights
  spec_kernel<<<dim3(1), dim3(256), 0, stream>>>(W, Om, wtT, omtT);
  // K2: bf16 conversions + Z0 = 0 (f32 copy lives in d_out)
  conv_kernel<<<dim3(2048), dim3(256), 0, stream>>>(adj, Hf, adjb, hfb, out, zbT0);
  // K3: pre = Hfeat * Omega^T + Q + bias
  pre_kernel<<<dim3((BB * NN) / MTP), dim3(256),
               MTP * HH * 2 + PANEL_ELEMS * 2, stream>>>(hfb, omtT, Q, bias, pre);

  // K4 x30: damped fixed-point iterations (ping-pong; iter 29 -> d_out)
  float*  zf[2] = { out, zf1 };
  bf16_t* zb[2] = { zbT0, zbT1 };
  const size_t iterLds =
      (size_t)MTI * NN * 2 + 2 * PANEL_ELEMS * 2 + (size_t)MTI * HH * 2; // 136KB
  int cur = 0;
  for (int it = 0; it < NITER; ++it) {
    const int nxt = cur ^ 1;
    iter_kernel<<<dim3(NN / MTI, BB), dim3(256), iterLds, stream>>>(
        adjb, zb[cur], zf[cur], wtT, pre, mask, zf[nxt], zb[nxt]);
    cur = nxt;
  }
}
